// HierarchicalBackmappingReadoutModule_73186242724417
// MI455X (gfx1250) — matrix operand order repaired and checked by reading, not compiled, not run on hardware
//
#include <hip/hip_runtime.h>
#include <hip/hip_bf16.h>

// ---------------- Types for WMMA ----------------
typedef __bf16 bf16_t;
typedef bf16_t v16bf __attribute__((ext_vector_type(16)));
typedef float  v8f   __attribute__((ext_vector_type(8)));

union ABu {
    uint4 q[2];
    v16bf b;
};

// ---------------- Constants ----------------
#define NATOMS 32768
#define INV    256
#define MUL    32
#define VOUT   4
#define HID    512
#define INV_OUT 2

__device__ __forceinline__ unsigned short f2bf(float f) {
    unsigned int u = __float_as_uint(f);
    unsigned int r = (u + 0x7FFFu + ((u >> 16) & 1u)) >> 16;   // RNE
    return (unsigned short)r;
}
__device__ __forceinline__ float bf2f(unsigned short h) {
    return __uint_as_float(((unsigned int)h) << 16);
}
__device__ __forceinline__ float silu_fast(float x) {
    // x * sigmoid(x) using v_exp_f32 + v_rcp_f32 (no IEEE divide chain)
    return x * __builtin_amdgcn_rcpf(1.f + __expf(-x));
}

// ---------------- Weight pack ----------------
// f32 row-major [K x Nout] -> fragment-major bf16-pair u32 array:
//   out[ (kt*Nout + n)*16 + rp ] = bf16(w[2r][n]) | bf16(w[2r+1][n])<<16
// where r = kt*16 + rp  (kt = K-tile of 32 rows = 16 row-pairs).
// A lane's 8 B-fragment dwords (rp0..rp0+3, rp0+8..rp0+11) form two contiguous runs -> 2x b128 loads.
__global__ void pack_bf16_frag(const float* __restrict__ w,
                               unsigned int* __restrict__ out,
                               int Khalf, int Nout) {
    int i = blockIdx.x * blockDim.x + threadIdx.x;
    int total = Khalf * Nout;
    if (i >= total) return;
    int r = i / Nout;
    int n = i - r * Nout;
    unsigned short lo = f2bf(w[(size_t)(2 * r) * Nout + n]);
    unsigned short hi = f2bf(w[(size_t)(2 * r + 1) * Nout + n]);
    int kt = r >> 4;
    int rp = r & 15;
    out[((size_t)kt * Nout + n) * 16 + rp] =
        (unsigned int)lo | ((unsigned int)hi << 16);
}

// ---------------- Wave-tiled WMMA GEMM stage ----------------
// A: LDS bf16 [32 x K] row-major. B: global fragment-major packed (see above).
// Each wave computes 32 x (16*NFRAG) of the output (2 M-frags x NFRAG N-frags).
template <int K, int NOUT, int NFRAG, bool SILU, bool OUTF32>
__device__ __forceinline__ void wave_gemm(const unsigned short* __restrict__ sA,
                                          const unsigned int* __restrict__ Bp,
                                          unsigned short* __restrict__ sOutBf,
                                          float* __restrict__ sOutF,
                                          int wave, int lane, float scale) {
    const int mrow  = lane & 15;
    const int khalf = (lane >> 4) << 3;  // 0 | 8 : lane-half K offset (elements)
    const int rp0   = (lane >> 4) << 2;  // 0 | 4 : row-pair base within k-tile

    const unsigned short* arow0 = &sA[mrow * K + khalf];
    const unsigned short* arow1 = &sA[(16 + mrow) * K + khalf];

    v8f C0[NFRAG], C1[NFRAG];
#pragma unroll
    for (int nf = 0; nf < NFRAG; ++nf) {
        C0[nf] = (v8f){0.f, 0.f, 0.f, 0.f, 0.f, 0.f, 0.f, 0.f};
        C1[nf] = (v8f){0.f, 0.f, 0.f, 0.f, 0.f, 0.f, 0.f, 0.f};
    }

    for (int k0 = 0; k0 < K; k0 += 32) {
        ABu a0, a1;
        // A fragment: two contiguous 16B runs per lane (K pairs 0..6 and 16..22)
        a0.q[0] = *(const uint4*)(&arow0[k0]);
        a0.q[1] = *(const uint4*)(&arow0[k0 + 16]);
        a1.q[0] = *(const uint4*)(&arow1[k0]);
        a1.q[1] = *(const uint4*)(&arow1[k0 + 16]);

        const int kt = k0 >> 5;
        if (k0 + 32 < K) {
            // prefetch next k-tile of this wave's B columns (L2-resident weights)
            __builtin_prefetch(
                Bp + ((size_t)(kt + 1) * NOUT + wave * (16 * NFRAG)) * 16, 0, 1);
        }
#pragma unroll
        for (int nf = 0; nf < NFRAG; ++nf) {
            const int col = wave * (16 * NFRAG) + nf * 16 + mrow;
            const unsigned int* bptr = Bp + ((size_t)kt * NOUT + col) * 16 + rp0;
            ABu bb;
            bb.q[0] = *(const uint4*)(bptr);
            bb.q[1] = *(const uint4*)(bptr + 8);
            C0[nf] = __builtin_amdgcn_wmma_f32_16x16x32_bf16(
                false, a0.b, false, bb.b, (short)0, C0[nf], false, false);
            C1[nf] = __builtin_amdgcn_wmma_f32_16x16x32_bf16(
                false, a1.b, false, bb.b, (short)0, C1[nf], false, false);
        }
    }

    // Epilogue: C/D layout: VGPR r, lanes 0-15 -> M=r, lanes 16-31 -> M=8+r, N=lane&15
    const int rowoff = (lane >> 4) << 3;
#pragma unroll
    for (int nf = 0; nf < NFRAG; ++nf) {
        const int col = wave * (16 * NFRAG) + nf * 16 + mrow;
#pragma unroll
        for (int r = 0; r < 8; ++r) {
            float v0 = C0[nf][r] * scale;
            float v1 = C1[nf][r] * scale;
            if (SILU) {
                v0 = silu_fast(v0);
                v1 = silu_fast(v1);
            }
            const int row0 = r + rowoff;
            const int row1 = 16 + r + rowoff;
            if (OUTF32) {
                sOutF[row0 * NOUT + col] = v0;
                sOutF[row1 * NOUT + col] = v1;
            } else {
                sOutBf[row0 * NOUT + col] = f2bf(v0);
                sOutBf[row1 * NOUT + col] = f2bf(v1);
            }
        }
    }
}

// ---------------- Fused per-32-atom pipeline ----------------
__global__ __launch_bounds__(256, 1) void fused_readout_kernel(
    const float* __restrict__ Xg,          // [N, 256]
    const float* __restrict__ eqf,         // [N, 32, 3]
    const int* __restrict__ types,         // [N, 1]
    const unsigned int* __restrict__ pW1L,
    const unsigned int* __restrict__ pW2L,
    const unsigned int* __restrict__ pW3L,
    const unsigned int* __restrict__ pW1R,
    const unsigned int* __restrict__ pW2R,
    const float* __restrict__ w3_ro,       // [512, 2] f32
    const float* __restrict__ bond_len,    // [32, 4, 1]
    float* __restrict__ out)               // inv_out [N,2] ++ eq [N,4,3]
{
    __shared__ unsigned short sX[32 * INV];   // 16 KB
    __shared__ unsigned short sH1[32 * HID];  // 32 KB
    __shared__ unsigned short sH2[32 * HID];  // 32 KB
    __shared__ float sW[32 * (MUL * VOUT)];   // 16 KB (per-atom hypernet weights)

    const int tid  = threadIdx.x;
    const int wave = tid >> 5;
    const int lane = tid & 31;
    const int blockAtom = blockIdx.x * 32;

    // Stage 0: load X tile, convert f32 -> bf16 into LDS (uniform trip count)
    for (int i = tid; i < 32 * INV; i += 256)
        sX[i] = f2bf(Xg[(size_t)blockAtom * INV + i]);
    __syncthreads();

    const float s256 = 0.0625f;                 // 1/sqrt(256)
    const float s512 = 0.04419417382415922f;    // 1/sqrt(512)

    // Hypernet (lat) chain: 256 -> 512 -> 512 -> 128
    wave_gemm<INV, HID, 4, true, false>(sX, pW1L, sH1, nullptr, wave, lane, s256);
    __syncthreads();
    wave_gemm<HID, HID, 4, true, false>(sH1, pW2L, sH2, nullptr, wave, lane, s512);
    __syncthreads();
    wave_gemm<HID, MUL * VOUT, 1, false, true>(sH2, pW3L, nullptr, sW, wave, lane, s512);
    __syncthreads();

    // Readout chain: 256 -> 512 -> 512 (layer3 is tiny, done on VALU below)
    wave_gemm<INV, HID, 4, true, false>(sX, pW1R, sH1, nullptr, wave, lane, s256);
    __syncthreads();
    wave_gemm<HID, HID, 4, true, false>(sH1, pW2R, sH2, nullptr, wave, lane, s512);
    __syncthreads();

    float* inv_out = out;                         // [N, 2]
    float* eq_out  = out + (size_t)NATOMS * 2;    // [N, 4, 3]

    if (tid < 128) {
        // equivariant contraction + normalize + bond-length scale
        const int aLoc = tid >> 2;
        const int v    = tid & 3;
        const int atom = blockAtom + aLoc;
        float e0 = 0.f, e1 = 0.f, e2 = 0.f;
#pragma unroll 4
        for (int u = 0; u < MUL; ++u) {
            const float w = sW[aLoc * (MUL * VOUT) + u * VOUT + v];
            const float* ef = eqf + ((size_t)atom * MUL + u) * 3;
            e0 += ef[0] * w;
            e1 += ef[1] * w;
            e2 += ef[2] * w;
        }
        const float rs32 = 0.1767766952966369f;   // 1/sqrt(32)
        e0 *= rs32; e1 *= rs32; e2 *= rs32;
        const float nrm = __builtin_sqrtf(e0 * e0 + e1 * e1 + e2 * e2);
        const float rin = 1.f / (nrm + 1e-10f);
        float o0 = e0 * rin, o1 = e1 * rin, o2 = e2 * rin;
        o0 = (o0 == o0) ? o0 : 0.f;                // nan_to_num
        o1 = (o1 == o1) ? o1 : 0.f;
        o2 = (o2 == o2) ? o2 : 0.f;
        const float bl = bond_len[types[atom] * VOUT + v];
        float* dst = eq_out + ((size_t)atom * VOUT + v) * 3;
        dst[0] = o0 * bl;
        dst[1] = o1 * bl;
        dst[2] = o2 * bl;
    } else if (tid < 192) {
        // readout layer 3: [32 x 512] @ [512 x 2]
        const int p    = tid - 128;
        const int aLoc = p >> 1;
        const int c    = p & 1;
        const int atom = blockAtom + aLoc;
        float s = 0.f;
#pragma unroll 8
        for (int k = 0; k < HID; ++k)
            s += bf2f(sH2[aLoc * HID + k]) * w3_ro[k * INV_OUT + c];
        inv_out[(size_t)atom * INV_OUT + c] = s * s512;
    }
}

// ---------------- Host launcher ----------------
extern "C" void kernel_launch(void* const* d_in, const int* in_sizes, int n_in,
                              void* d_out, int out_size, void* d_ws, size_t ws_size,
                              hipStream_t stream) {
    (void)in_sizes; (void)n_in; (void)out_size; (void)ws_size;

    const float* inv_features = (const float*)d_in[0];
    const float* eq_features  = (const float*)d_in[1];
    const int*   atom_types   = (const int*)d_in[2];
    const float* w1_lat       = (const float*)d_in[3];
    const float* w2_lat       = (const float*)d_in[4];
    const float* w3_lat       = (const float*)d_in[5];
    const float* w1_ro        = (const float*)d_in[6];
    const float* w2_ro        = (const float*)d_in[7];
    const float* w3_ro        = (const float*)d_in[8];
    const float* bond_lengths = (const float*)d_in[9];
    float* out = (float*)d_out;

    // Workspace layout (u32 elements)
    unsigned int* ws   = (unsigned int*)d_ws;
    unsigned int* pW1L = ws;                  // 128*512 = 65536
    unsigned int* pW2L = pW1L + 65536;        // 256*512 = 131072
    unsigned int* pW3L = pW2L + 131072;       // 256*128 = 32768
    unsigned int* pW1R = pW3L + 32768;        // 65536
    unsigned int* pW2R = pW1R + 65536;        // 131072

    auto pk = [&](const float* w, unsigned int* o, int Kh, int Nn) {
        int total = Kh * Nn;
        pack_bf16_frag<<<(total + 255) / 256, 256, 0, stream>>>(w, o, Kh, Nn);
    };
    pk(w1_lat, pW1L, INV / 2, HID);
    pk(w2_lat, pW2L, HID / 2, HID);
    pk(w3_lat, pW3L, HID / 2, MUL * VOUT);
    pk(w1_ro,  pW1R, INV / 2, HID);
    pk(w2_ro,  pW2R, HID / 2, HID);

    fused_readout_kernel<<<NATOMS / 32, 256, 0, stream>>>(
        inv_features, eq_features, atom_types,
        pW1L, pW2L, pW3L, pW1R, pW2R,
        w3_ro, bond_lengths, out);
}
